// MolEncoder_42666205118513
// MI455X (gfx1250) — compile-verified, matrix operand order checked by reading, not run on hardware
//
#include <hip/hip_runtime.h>
#include <hip/hip_bf16.h>

typedef __attribute__((ext_vector_type(16))) __bf16 v16bf;
typedef __attribute__((ext_vector_type(8)))  float  v8f;

#define DEVFN __device__ __forceinline__

constexpr int N_NODES = 20000;
constexpr int N_EDGES = 40000;
constexpr int N_GRAPH = 1000;

// ---- WMMA fragment helpers (wave32, 16x16x32 bf16, per CDNA5 ISA layouts) ----
DEVFN int frag_k(int j, int lane) {
    // A/B 16-bit K packing: j<8 -> K=j ; j>=8 -> K=16+(j-8) ; lanes 16..31 add 8
    return (j & 7) + ((j >> 3) << 4) + ((lane >> 4) << 3);
}

// B matrices are pre-swizzled to fragment order: [ktile][ntile][lane][16 bf16],
// so a lane's whole fragment is one contiguous 32-byte load (2 x b128).
DEVFN v16bf load_bfrag_sw(const __bf16* Bs, int ktile, int NT, int nt, int lane) {
    return *(const v16bf*)(Bs + ((size_t)((ktile * NT + nt) * 32 + lane) << 4));
}

// A fragment from an f32 row-major LDS tile: two runs of 8 consecutive floats.
DEVFN v16bf make_afrag_f32(const float* U, int ldk, int k0, int lane) {
    int m = lane & 15;
    const float* p = U + m * ldk + k0 + ((lane >> 4) << 3);
    float4 x0 = *(const float4*)(p);
    float4 x1 = *(const float4*)(p + 4);
    float4 x2 = *(const float4*)(p + 16);
    float4 x3 = *(const float4*)(p + 20);
    v16bf a;
    a[0] = (__bf16)x0.x; a[1] = (__bf16)x0.y; a[2] = (__bf16)x0.z; a[3] = (__bf16)x0.w;
    a[4] = (__bf16)x1.x; a[5] = (__bf16)x1.y; a[6] = (__bf16)x1.z; a[7] = (__bf16)x1.w;
    a[8] = (__bf16)x2.x; a[9] = (__bf16)x2.y; a[10] = (__bf16)x2.z; a[11] = (__bf16)x2.w;
    a[12] = (__bf16)x3.x; a[13] = (__bf16)x3.y; a[14] = (__bf16)x3.z; a[15] = (__bf16)x3.w;
    return a;
}

// A fragment from 4 register-resident float4 runs (optionally scaled).
DEVFN v16bf frag_from_raw(float4 r0, float4 r1, float4 r2, float4 r3, float s) {
    v16bf a;
    a[0] = (__bf16)(s * r0.x); a[1] = (__bf16)(s * r0.y); a[2] = (__bf16)(s * r0.z); a[3] = (__bf16)(s * r0.w);
    a[4] = (__bf16)(s * r1.x); a[5] = (__bf16)(s * r1.y); a[6] = (__bf16)(s * r1.z); a[7] = (__bf16)(s * r1.w);
    a[8] = (__bf16)(s * r2.x); a[9] = (__bf16)(s * r2.y); a[10] = (__bf16)(s * r2.z); a[11] = (__bf16)(s * r2.w);
    a[12] = (__bf16)(s * r3.x); a[13] = (__bf16)(s * r3.y); a[14] = (__bf16)(s * r3.z); a[15] = (__bf16)(s * r3.w);
    return a;
}
DEVFN v16bf frag_from_raw_ns(float4 r0, float4 r1, float4 r2, float4 r3) {
    v16bf a;
    a[0] = (__bf16)r0.x; a[1] = (__bf16)r0.y; a[2] = (__bf16)r0.z; a[3] = (__bf16)r0.w;
    a[4] = (__bf16)r1.x; a[5] = (__bf16)r1.y; a[6] = (__bf16)r1.z; a[7] = (__bf16)r1.w;
    a[8] = (__bf16)r2.x; a[9] = (__bf16)r2.y; a[10] = (__bf16)r2.z; a[11] = (__bf16)r2.w;
    a[12] = (__bf16)r3.x; a[13] = (__bf16)r3.y; a[14] = (__bf16)r3.z; a[15] = (__bf16)r3.w;
    return a;
}

DEVFN v8f wmma_bf16(v16bf a, v16bf b, v8f c) {
    return __builtin_amdgcn_wmma_f32_16x16x32_bf16(false, a, false, b, (short)0, c, false, false);
}

DEVFN float sigm(float x) { return 1.0f / (1.0f + __expf(-x)); }

DEVFN void atomicMaxF(float* addr, float v) {
    unsigned int* ai = (unsigned int*)addr;
    unsigned int cur = *ai;
    while (__uint_as_float(cur) < v) {
        unsigned int prev = atomicCAS(ai, cur, __float_as_uint(v));
        if (prev == cur) break;
        cur = prev;
    }
}

// ---------------- weight packing (f32 -> bf16 plain, then fragment swizzle) ----------------
__global__ void pack_copy_kernel(const float* s, __bf16* d, int n) {
    int i = blockIdx.x * blockDim.x + threadIdx.x;
    if (i < n) d[i] = (__bf16)s[i];
}
__global__ void pack_transpose_kernel(const float* s, __bf16* d, int K, int NC, int ldn) {
    int i = blockIdx.x * blockDim.x + threadIdx.x;
    if (i >= K * NC) return;
    int k = i / NC, n = i % NC;
    d[k * ldn + n] = (__bf16)s[n * K + k];
}
__global__ void pack_wlin_kernel(const float* W, __bf16* d) {
    int i = blockIdx.x * blockDim.x + threadIdx.x;
    if (i >= 96 * 64) return;
    int k = i >> 6, n = i & 63;
    d[i] = (k < 74) ? (__bf16)W[k * 64 + n] : (__bf16)0.0f;
}
// B for fused edge GEMM: rows k = f*64+h (f=0..11 from W_edge, f=12 = bias block), cols o
__global__ void pack_wbe_kernel(const float* We, const float* be, __bf16* d) {
    int i = blockIdx.x * blockDim.x + threadIdx.x;
    if (i >= 832 * 64) return;
    int k = i >> 6, n = i & 63;
    int f = k >> 6, hh = k & 63;
    float v = (f < 12) ? We[f * 4096 + hh * 64 + n] : be[hh * 64 + n];
    d[i] = (__bf16)v;
}
// plain [K][NC] bf16 -> fragment order [ktile][ntile][lane][16]
__global__ void swizzle_b_kernel(const __bf16* src, __bf16* dst, int K, int NC) {
    int i = blockIdx.x * blockDim.x + threadIdx.x;
    if (i >= K * NC) return;
    int j = i & 15;
    int lane = (i >> 4) & 31;
    int NT = NC >> 4;
    int nt = (i >> 9) % NT;
    int kk = i / (NT << 9);
    int k = kk * 32 + frag_k(j, lane);
    int n = nt * 16 + (lane & 15);
    dst[i] = src[k * NC + n];
}

__global__ void fill_kernel(float* p, float v, int n) {
    int i = blockIdx.x * blockDim.x + threadIdx.x;
    if (i < n) p[i] = v;
}

// ---------------- h0 = relu(x @ W_lin + b_lin); also initializes hid ----------------
__global__ void h0_kernel(const float* __restrict__ x, const __bf16* __restrict__ Wlin_s,
                          const float* __restrict__ b_lin, float* __restrict__ h0,
                          float* __restrict__ hid) {
    extern __shared__ char smem[];
    __bf16* sW = (__bf16*)smem;                 // 96*64 bf16, swizzled
    int tid = threadIdx.x, lane = tid & 31, wave = tid >> 5;
    for (int i = tid; i < 96 * 64; i += blockDim.x) sW[i] = Wlin_s[i];
    __syncthreads();
    float* sX = (float*)(smem + 12288) + wave * (16 * 96);
    int m = lane & 15, mb = (lane >> 4) << 3;
    int tiles = N_NODES / 16;
    int wstride = gridDim.x * (blockDim.x >> 5);
    for (int tile = blockIdx.x * (blockDim.x >> 5) + wave; tile < tiles; tile += wstride) {
        int rbase = tile * 16;
        for (int r = 0; r < 16; ++r)
            for (int c = lane; c < 96; c += 32)
                sX[r * 96 + c] = (c < 74) ? x[(rbase + r) * 74 + c] : 0.0f;
        v8f acc[4]; { v8f z = {}; for (int t = 0; t < 4; ++t) acc[t] = z; }
        for (int kk = 0; kk < 3; ++kk) {
            v16bf a = make_afrag_f32(sX, 96, kk * 32, lane);
#pragma unroll
            for (int t = 0; t < 4; ++t)
                acc[t] = wmma_bf16(a, load_bfrag_sw(sW, kk, 4, t, lane), acc[t]);
        }
#pragma unroll
        for (int t = 0; t < 4; ++t) {
            int col = t * 16 + m;
            float b = b_lin[col];
#pragma unroll
            for (int i = 0; i < 8; ++i) {
                float v = acc[t][i] + b;
                v = v > 0.0f ? v : 0.0f;
                int row = rbase + mb + i;
                h0[row * 64 + col] = v;
                hid[row * 64 + col] = v;
            }
        }
    }
}

// ------- fused NNConv message: msg = (ea ⊗ h_src) @ Wfho, scatter-add into aggr -------
// A operands live entirely in registers: lane L needs only 32 of the 64 h values of
// row (L&15) (lanes m and m+16 cover disjoint halves), reused across all 13 f-blocks.
__global__ void edge_msg_kernel(const float* __restrict__ hid, const int* __restrict__ src,
                                const int* __restrict__ dst, const float* __restrict__ edge_attr,
                                const __bf16* __restrict__ WBe_s, float* __restrict__ aggr) {
    extern __shared__ char smem[];
    __bf16* sB = (__bf16*)smem;                 // 832*64 bf16 = 104KB, swizzled
    int tid = threadIdx.x, lane = tid & 31, wave = tid >> 5;
    for (int i = tid; i < 832 * 64; i += blockDim.x) sB[i] = WBe_s[i];
    __syncthreads();
    int m = lane & 15, hi = lane >> 4, mb = hi << 3;
    int tiles = N_EDGES / 16;
    int wstride = gridDim.x * (blockDim.x >> 5);
    for (int tile = blockIdx.x * (blockDim.x >> 5) + wave; tile < tiles; tile += wstride) {
        int ebase = tile * 16;
        int srow = src[ebase + m];
        const float4* hp = (const float4*)(hid + srow * 64);
        float4 raw[8];
#pragma unroll
        for (int g = 0; g < 4; ++g) {
            raw[2 * g]     = hp[g * 4 + hi * 2];
            raw[2 * g + 1] = hp[g * 4 + hi * 2 + 1];
        }
        const float4* ep = (const float4*)(edge_attr + (ebase + m) * 12);
        float4 e0 = ep[0], e1 = ep[1], e2 = ep[2];
        float eav_arr[13] = {e0.x, e0.y, e0.z, e0.w, e1.x, e1.y, e1.z, e1.w,
                             e2.x, e2.y, e2.z, e2.w, 1.0f};
        v8f acc[4]; { v8f z = {}; for (int t = 0; t < 4; ++t) acc[t] = z; }
#pragma unroll
        for (int kk = 0; kk < 26; ++kk) {       // K = 13*64 = 832 (f block = kk>>1)
            int f = kk >> 1, half = kk & 1;
            float s = eav_arr[f];
            v16bf a = frag_from_raw(raw[4 * half], raw[4 * half + 1],
                                    raw[4 * half + 2], raw[4 * half + 3], s);
#pragma unroll
            for (int t = 0; t < 4; ++t)
                acc[t] = wmma_bf16(a, load_bfrag_sw(sB, kk, 4, t, lane), acc[t]);
        }
#pragma unroll
        for (int t = 0; t < 4; ++t) {
            int col = t * 16 + m;
#pragma unroll
            for (int i = 0; i < 8; ++i) {
                int e = ebase + mb + i;
                atomicAdd(&aggr[dst[e] * 64 + col], acc[t][i]);
            }
        }
    }
}

// ------- fused root transform + relu + single-step GRU (in-place update of hid) -------
__global__ void gru_kernel(float* __restrict__ hid, const float* __restrict__ aggr,
                           const __bf16* __restrict__ Wroot_s, const __bf16* __restrict__ WihT_s,
                           const __bf16* __restrict__ WhhT_s, const float* __restrict__ b_conv,
                           const float* __restrict__ b_ih, const float* __restrict__ b_hh) {
    extern __shared__ char smem[];
    __bf16* sWr = (__bf16*)smem;                // 64*64  swizzled (NT=4)
    __bf16* sWi = sWr + 4096;                   // 64*192 swizzled (NT=12)
    __bf16* sWh = sWi + 12288;                  // 64*192 swizzled (NT=12)
    int tid = threadIdx.x, lane = tid & 31, wave = tid >> 5;
    for (int i = tid; i < 4096; i += blockDim.x) sWr[i] = Wroot_s[i];
    for (int i = tid; i < 12288; i += blockDim.x) { sWi[i] = WihT_s[i]; sWh[i] = WhhT_s[i]; }
    __syncthreads();
    float* sM = (float*)(smem + 57344) + wave * 1024;   // 16x64 m tile (layout transpose)
    int m = lane & 15, hi = lane >> 4, mb = hi << 3;
    int tiles = N_NODES / 16;
    int wstride = gridDim.x * (blockDim.x >> 5);
    for (int tile = blockIdx.x * (blockDim.x >> 5) + wave; tile < tiles; tile += wstride) {
        int rbase = tile * 16;
        // h fragments straight from global (register-resident raw values)
        const float4* hp = (const float4*)(hid + (rbase + m) * 64);
        float4 raw[8];
#pragma unroll
        for (int g = 0; g < 4; ++g) {
            raw[2 * g]     = hp[g * 4 + hi * 2];
            raw[2 * g + 1] = hp[g * 4 + hi * 2 + 1];
        }
        v16bf ah0 = frag_from_raw_ns(raw[0], raw[1], raw[2], raw[3]);
        v16bf ah1 = frag_from_raw_ns(raw[4], raw[5], raw[6], raw[7]);
        // u = h @ W_root
        v8f cu[4];
#pragma unroll
        for (int t = 0; t < 4; ++t) {
            v8f c = {};
            c = wmma_bf16(ah0, load_bfrag_sw(sWr, 0, 4, t, lane), c);
            c = wmma_bf16(ah1, load_bfrag_sw(sWr, 1, 4, t, lane), c);
            cu[t] = c;
        }
        // m = relu(aggr + u + b_conv) -> LDS (D-layout write, A-layout read)
#pragma unroll
        for (int t = 0; t < 4; ++t) {
            int col = t * 16 + m;
            float bc = b_conv[col];
#pragma unroll
            for (int i = 0; i < 8; ++i) {
                int row = mb + i;
                float v = cu[t][i] + aggr[(rbase + row) * 64 + col] + bc;
                sM[row * 64 + col] = v > 0.0f ? v : 0.0f;
            }
        }
        v16bf am0 = make_afrag_f32(sM, 64, 0, lane);
        v16bf am1 = make_afrag_f32(sM, 64, 32, lane);
        float fr[4][8], fz[4][8];
        // r gate (chunk 0), z gate (chunk 1)
#pragma unroll
        for (int ch = 0; ch < 2; ++ch) {
#pragma unroll
            for (int t = 0; t < 4; ++t) {
                int nt = ch * 4 + t;
                v8f gi = {}, gh = {};
                gi = wmma_bf16(am0, load_bfrag_sw(sWi, 0, 12, nt, lane), gi);
                gi = wmma_bf16(am1, load_bfrag_sw(sWi, 1, 12, nt, lane), gi);
                gh = wmma_bf16(ah0, load_bfrag_sw(sWh, 0, 12, nt, lane), gh);
                gh = wmma_bf16(ah1, load_bfrag_sw(sWh, 1, 12, nt, lane), gh);
                int c2 = t * 16 + m;
                float bi = b_ih[ch * 64 + c2], bh = b_hh[ch * 64 + c2];
#pragma unroll
                for (int i = 0; i < 8; ++i) {
                    float s = sigm(gi[i] + bi + gh[i] + bh);
                    if (ch == 0) fr[t][i] = s; else fz[t][i] = s;
                }
            }
        }
        // n gate (chunk 2) + combine + store
#pragma unroll
        for (int t = 0; t < 4; ++t) {
            int nt = 8 + t;
            v8f gi = {}, gh = {};
            gi = wmma_bf16(am0, load_bfrag_sw(sWi, 0, 12, nt, lane), gi);
            gi = wmma_bf16(am1, load_bfrag_sw(sWi, 1, 12, nt, lane), gi);
            gh = wmma_bf16(ah0, load_bfrag_sw(sWh, 0, 12, nt, lane), gh);
            gh = wmma_bf16(ah1, load_bfrag_sw(sWh, 1, 12, nt, lane), gh);
            int c2 = t * 16 + m;
            float bi = b_ih[128 + c2], bh = b_hh[128 + c2];
#pragma unroll
            for (int i = 0; i < 8; ++i) {
                float nn = tanhf(gi[i] + bi + fr[t][i] * (gh[i] + bh));
                float hold = hid[(rbase + mb + i) * 64 + c2];
                float z = fz[t][i];
                hid[(rbase + mb + i) * 64 + c2] = (1.0f - z) * nn + z * hold;
            }
        }
    }
}

__global__ void feat_kernel(const float* h0, const float* hid, float* feat) {
    int i = blockIdx.x * blockDim.x + threadIdx.x;
    if (i >= N_NODES * 128) return;
    int n = i >> 7, c = i & 127;
    feat[i] = (c < 64) ? h0[n * 64 + c] : hid[n * 64 + (c - 64)];
}

// ------- Set2Set LSTM pre-activations g = [qstar|hl] @ Wcat + biases (K=384) -------
__global__ void lstm_g_kernel(const float* __restrict__ qstar, const float* __restrict__ hl,
                              const __bf16* __restrict__ Wcat_s, const float* __restrict__ b_ihl,
                              const float* __restrict__ b_hhl, float* __restrict__ g) {
    extern __shared__ char smem[];
    float* U = (float*)smem;                    // 16x384
    int tid = threadIdx.x, lane = tid & 31, wave = tid >> 5;
    int rbase = blockIdx.x * 16;
    for (int i = tid; i < 16 * 384; i += blockDim.x) {
        int r = i / 384, k = i % 384;
        int row = rbase + r;
        float v = 0.0f;
        if (row < N_GRAPH) v = (k < 256) ? qstar[row * 256 + k] : hl[row * 128 + (k - 256)];
        U[i] = v;
    }
    __syncthreads();
    int m = lane & 15, mb = (lane >> 4) << 3;
    v8f acc[4]; { v8f z = {}; for (int t = 0; t < 4; ++t) acc[t] = z; }
    for (int kk = 0; kk < 12; ++kk) {
        v16bf a = make_afrag_f32(U, 384, kk * 32, lane);
#pragma unroll
        for (int t = 0; t < 4; ++t)
            acc[t] = wmma_bf16(a, load_bfrag_sw(Wcat_s, kk, 32, wave * 4 + t, lane), acc[t]);
    }
#pragma unroll
    for (int t = 0; t < 4; ++t) {
        int col = (wave * 4 + t) * 16 + m;
        float bb = b_ihl[col] + b_hhl[col];
#pragma unroll
        for (int i = 0; i < 8; ++i) {
            int row = rbase + mb + i;
            if (row < N_GRAPH) g[row * 512 + col] = acc[t][i] + bb;
        }
    }
}

__global__ void lstm_gate_kernel(const float* g, float* hl, float* cl) {
    int idx = blockIdx.x * blockDim.x + threadIdx.x;
    if (idx >= N_GRAPH * 128) return;
    int b = idx >> 7, c = idx & 127;
    const float* gr = g + b * 512;
    float gi = gr[c], gf = gr[128 + c], gg = gr[256 + c], go = gr[384 + c];
    float cn = sigm(gf) * cl[idx] + sigm(gi) * tanhf(gg);
    cl[idx] = cn;
    hl[idx] = sigm(go) * tanhf(cn);
}

__global__ void att_e_kernel(const float* __restrict__ feat, const float* __restrict__ hl,
                             const int* __restrict__ batch, float* __restrict__ e,
                             float* __restrict__ emax) {
    int n = blockIdx.x * blockDim.x + threadIdx.x;
    if (n >= N_NODES) return;
    int b = batch[n];
    const float4* f = (const float4*)(feat + n * 128);
    const float4* q = (const float4*)(hl + b * 128);
    float acc = 0.0f;
    for (int k = 0; k < 32; ++k) {
        float4 a = f[k], c = q[k];
        acc += a.x * c.x + a.y * c.y + a.z * c.z + a.w * c.w;
    }
    e[n] = acc;
    atomicMaxF(&emax[b], acc);
}

__global__ void att_sum_kernel(const float* e, const int* batch, const float* emax,
                               float* aun, float* denom) {
    int n = blockIdx.x * blockDim.x + threadIdx.x;
    if (n >= N_NODES) return;
    int b = batch[n];
    float em = emax[b];
    if (!(em > -1e30f)) em = 0.0f;      // empty-segment -inf -> 0 per reference
    float a = expf(e[n] - em);
    aun[n] = a;
    atomicAdd(&denom[b], a);
}

__global__ void att_r_kernel(const float* aun, const float* denom, const int* batch,
                             const float* feat, float* rvec) {
    int i = blockIdx.x * blockDim.x + threadIdx.x;
    if (i >= N_NODES * 128) return;
    int n = i >> 7, k = i & 127;
    int b = batch[n];
    float w = aun[n] / (denom[b] + 1e-16f);
    atomicAdd(&rvec[b * 128 + k], w * feat[i]);
}

__global__ void qstar_kernel(const float* hl, const float* rvec, float* qstar) {
    int i = blockIdx.x * blockDim.x + threadIdx.x;
    if (i >= N_GRAPH * 256) return;
    int b = i >> 8, k = i & 255;
    qstar[i] = (k < 128) ? hl[b * 128 + k] : rvec[b * 128 + (k - 128)];
}

// ------- final projection + PReLU -------
__global__ void out_kernel(const float* __restrict__ qstar, const __bf16* __restrict__ Wout_s,
                           const float* __restrict__ b_out, const float* __restrict__ prelu_a,
                           float* __restrict__ y) {
    extern __shared__ char smem[];
    float* U = (float*)smem;                    // 16x256
    int tid = threadIdx.x, lane = tid & 31, wave = tid >> 5;
    int rbase = blockIdx.x * 16;
    for (int i = tid; i < 16 * 256; i += blockDim.x) {
        int r = i / 256, k = i % 256;
        int row = rbase + r;
        U[i] = (row < N_GRAPH) ? qstar[row * 256 + k] : 0.0f;
    }
    __syncthreads();
    int m = lane & 15, mb = (lane >> 4) << 3;
    v8f acc[4]; { v8f z = {}; for (int t = 0; t < 4; ++t) acc[t] = z; }
    for (int kk = 0; kk < 8; ++kk) {
        v16bf a = make_afrag_f32(U, 256, kk * 32, lane);
#pragma unroll
        for (int t = 0; t < 4; ++t)
            acc[t] = wmma_bf16(a, load_bfrag_sw(Wout_s, kk, 32, wave * 4 + t, lane), acc[t]);
    }
    float al = prelu_a[0];
#pragma unroll
    for (int t = 0; t < 4; ++t) {
        int col = (wave * 4 + t) * 16 + m;
        float bb = b_out[col];
#pragma unroll
        for (int i = 0; i < 8; ++i) {
            int row = rbase + mb + i;
            if (row < N_GRAPH) {
                float v = acc[t][i] + bb;
                y[row * 512 + col] = (v >= 0.0f) ? v : al * v;
            }
        }
    }
}

static inline int cdiv(int a, int b) { return (a + b - 1) / b; }

extern "C" void kernel_launch(void* const* d_in, const int* in_sizes, int n_in,
                              void* d_out, int out_size, void* d_ws, size_t ws_size,
                              hipStream_t stream) {
    (void)in_sizes; (void)n_in; (void)out_size; (void)ws_size;
    const float* x         = (const float*)d_in[0];
    const float* edge_attr = (const float*)d_in[1];
    const int*   eidx      = (const int*)d_in[2];
    const int*   src = eidx;
    const int*   dst = eidx + N_EDGES;
    const int*   batch     = (const int*)d_in[3];
    const float* W_lin  = (const float*)d_in[4];
    const float* b_lin  = (const float*)d_in[5];
    const float* W_edge = (const float*)d_in[6];
    const float* b_edge = (const float*)d_in[7];
    const float* W_root = (const float*)d_in[8];
    const float* b_conv = (const float*)d_in[9];
    const float* W_ih   = (const float*)d_in[10];
    const float* W_hh   = (const float*)d_in[11];
    const float* b_ih   = (const float*)d_in[12];
    const float* b_hh   = (const float*)d_in[13];
    const float* W_ihl  = (const float*)d_in[14];
    const float* W_hhl  = (const float*)d_in[15];
    const float* b_ihl  = (const float*)d_in[16];
    const float* b_hhl  = (const float*)d_in[17];
    const float* W_out  = (const float*)d_in[18];
    const float* b_out  = (const float*)d_in[19];
    const float* prelu  = (const float*)d_in[20];
    float* y = (float*)d_out;

    char* ws = (char*)d_ws;
    size_t off = 0;
    auto alloc = [&](size_t bytes) -> void* {
        void* p = ws + off;
        off = (off + bytes + 255) & ~(size_t)255;
        return p;
    };
    float* h0f   = (float*)alloc((size_t)N_NODES * 64 * 4);
    float* hidf  = (float*)alloc((size_t)N_NODES * 64 * 4);
    float* aggrf = (float*)alloc((size_t)N_NODES * 64 * 4);
    float* featf = (float*)alloc((size_t)N_NODES * 128 * 4);
    float* gf    = (float*)alloc((size_t)N_GRAPH * 512 * 4);
    float* qsf   = (float*)alloc((size_t)N_GRAPH * 256 * 4);
    float* hlf   = (float*)alloc((size_t)N_GRAPH * 128 * 4);
    float* clf   = (float*)alloc((size_t)N_GRAPH * 128 * 4);
    float* ef    = (float*)alloc((size_t)N_NODES * 4);
    float* aunf  = (float*)alloc((size_t)N_NODES * 4);
    float* emaxf = (float*)alloc((size_t)N_GRAPH * 4);
    float* denf  = (float*)alloc((size_t)N_GRAPH * 4);
    float* rvecf = (float*)alloc((size_t)N_GRAPH * 128 * 4);
    __bf16* wlin_s  = (__bf16*)alloc((size_t)96 * 64 * 2);
    __bf16* wbe_s   = (__bf16*)alloc((size_t)832 * 64 * 2);
    __bf16* wroot_s = (__bf16*)alloc((size_t)64 * 64 * 2);
    __bf16* wihT_s  = (__bf16*)alloc((size_t)64 * 192 * 2);
    __bf16* whhT_s  = (__bf16*)alloc((size_t)64 * 192 * 2);
    __bf16* wcat_s  = (__bf16*)alloc((size_t)384 * 512 * 2);
    __bf16* wout_s  = (__bf16*)alloc((size_t)256 * 512 * 2);
    __bf16* tmp     = (__bf16*)alloc((size_t)384 * 512 * 2);   // plain staging (largest)

    // ---- pack weights to bf16 plain layout, then swizzle to fragment order ----
    pack_wlin_kernel<<<cdiv(96 * 64, 256), 256, 0, stream>>>(W_lin, tmp);
    swizzle_b_kernel<<<cdiv(96 * 64, 256), 256, 0, stream>>>(tmp, wlin_s, 96, 64);

    pack_wbe_kernel<<<cdiv(832 * 64, 256), 256, 0, stream>>>(W_edge, b_edge, tmp);
    swizzle_b_kernel<<<cdiv(832 * 64, 256), 256, 0, stream>>>(tmp, wbe_s, 832, 64);

    pack_copy_kernel<<<cdiv(64 * 64, 256), 256, 0, stream>>>(W_root, tmp, 64 * 64);
    swizzle_b_kernel<<<cdiv(64 * 64, 256), 256, 0, stream>>>(tmp, wroot_s, 64, 64);

    pack_transpose_kernel<<<cdiv(64 * 192, 256), 256, 0, stream>>>(W_ih, tmp, 64, 192, 192);
    swizzle_b_kernel<<<cdiv(64 * 192, 256), 256, 0, stream>>>(tmp, wihT_s, 64, 192);

    pack_transpose_kernel<<<cdiv(64 * 192, 256), 256, 0, stream>>>(W_hh, tmp, 64, 192, 192);
    swizzle_b_kernel<<<cdiv(64 * 192, 256), 256, 0, stream>>>(tmp, whhT_s, 64, 192);

    pack_transpose_kernel<<<cdiv(256 * 512, 256), 256, 0, stream>>>(W_ihl, tmp, 256, 512, 512);
    pack_transpose_kernel<<<cdiv(128 * 512, 256), 256, 0, stream>>>(W_hhl, tmp + 256 * 512, 128, 512, 512);
    swizzle_b_kernel<<<cdiv(384 * 512, 256), 256, 0, stream>>>(tmp, wcat_s, 384, 512);

    pack_copy_kernel<<<cdiv(256 * 512, 256), 256, 0, stream>>>(W_out, tmp, 256 * 512);
    swizzle_b_kernel<<<cdiv(256 * 512, 256), 256, 0, stream>>>(tmp, wout_s, 256, 512);

    // ---- h0 / hid init ----
    h0_kernel<<<160, 128, 36864, stream>>>(x, wlin_s, b_lin, h0f, hidf);

    // ---- 3 message-passing steps ----
    for (int s = 0; s < 3; ++s) {
        fill_kernel<<<cdiv(N_NODES * 64, 256), 256, 0, stream>>>(aggrf, 0.0f, N_NODES * 64);
        edge_msg_kernel<<<320, 128, 106496, stream>>>(hidf, src, dst, edge_attr, wbe_s, aggrf);
        gru_kernel<<<160, 128, 73728, stream>>>(hidf, aggrf, wroot_s, wihT_s, whhT_s,
                                                b_conv, b_ih, b_hh);
    }

    feat_kernel<<<cdiv(N_NODES * 128, 256), 256, 0, stream>>>(h0f, hidf, featf);

    // ---- Set2Set ----
    fill_kernel<<<cdiv(N_GRAPH * 256, 256), 256, 0, stream>>>(qsf, 0.0f, N_GRAPH * 256);
    fill_kernel<<<cdiv(N_GRAPH * 128, 256), 256, 0, stream>>>(hlf, 0.0f, N_GRAPH * 128);
    fill_kernel<<<cdiv(N_GRAPH * 128, 256), 256, 0, stream>>>(clf, 0.0f, N_GRAPH * 128);
    int gtiles = cdiv(N_GRAPH, 16);
    for (int s = 0; s < 3; ++s) {
        lstm_g_kernel<<<gtiles, 256, 24576, stream>>>(qsf, hlf, wcat_s, b_ihl, b_hhl, gf);
        lstm_gate_kernel<<<cdiv(N_GRAPH * 128, 256), 256, 0, stream>>>(gf, hlf, clf);
        fill_kernel<<<cdiv(N_GRAPH, 256), 256, 0, stream>>>(emaxf, -__builtin_inff(), N_GRAPH);
        fill_kernel<<<cdiv(N_GRAPH, 256), 256, 0, stream>>>(denf, 0.0f, N_GRAPH);
        fill_kernel<<<cdiv(N_GRAPH * 128, 256), 256, 0, stream>>>(rvecf, 0.0f, N_GRAPH * 128);
        att_e_kernel<<<cdiv(N_NODES, 256), 256, 0, stream>>>(featf, hlf, batch, ef, emaxf);
        att_sum_kernel<<<cdiv(N_NODES, 256), 256, 0, stream>>>(ef, batch, emaxf, aunf, denf);
        att_r_kernel<<<cdiv(N_NODES * 128, 256), 256, 0, stream>>>(aunf, denf, batch, featf, rvecf);
        qstar_kernel<<<cdiv(N_GRAPH * 256, 256), 256, 0, stream>>>(hlf, rvecf, qsf);
    }

    // ---- output projection + PReLU ----
    out_kernel<<<gtiles, 256, 16384, stream>>>(qsf, wout_s, b_out, prelu, y);
}
